// Net_61108794688241
// MI455X (gfx1250) — compile-verified, matrix operand order checked by reading, not run on hardware
//
#include <hip/hip_runtime.h>
#include <hip/hip_bf16.h>
#include <math.h>

typedef __attribute__((ext_vector_type(16))) _Float16 v16h;
typedef __attribute__((ext_vector_type(8)))  _Float16 v8h;
typedef __attribute__((ext_vector_type(8)))  float    v8f;
typedef __attribute__((ext_vector_type(4)))  unsigned int v4u;
typedef __attribute__((ext_vector_type(8)))  int v8i;
typedef __attribute__((ext_vector_type(4)))  int v4i;

#define B_   32
#define T_   4096
#define D_   80
#define DH   78        // high-frequency dims
#define DHP  96        // padded to 3*32 (WMMA K)
#define KH   1000      // high clusters
#define KHP  1008      // padded to 63*16
#define NT   63        // N tiles of 16
#define KL   50        // low clusters
#define NROWS (B_*T_)  // 131072

#define BFRAG_HALVES (NT * 3 * 32 * 16)   // 96768 halves = 193536 B
#define BFRAG_BYTES  (BFRAG_HALVES * 2)

// ---- workspace layout (bytes, 256-aligned chunks) ----
#define WS_BFRAG   0                        // 193536 B
#define WS_INVH    193536                   // 1008 f32 (padded to 4096 B)
#define WS_INVL    (193536 + 4096)          // 64 f32
#define WS_IDXH    (193536 + 4096 + 256)    // 131072 i32

// ---- output layout (floats) ----
#define OUT_IDXH   (NROWS * D_)             // 10485760
#define OUT_IDXL   (OUT_IDXH + NROWS)

__global__ __launch_bounds__(256) void prep_norms(const float* __restrict__ wh,
                                                  const float* __restrict__ wl,
                                                  float* __restrict__ invh,
                                                  float* __restrict__ invl) {
  int k = blockIdx.x * 256 + threadIdx.x;
  if (k < KHP) {
    float inv = 0.0f;
    if (k < KH) {
      float s = 0.0f;
      for (int d = 0; d < DH; ++d) { float w = wh[k * DH + d]; s += w * w; }
      inv = 1.0f / sqrtf(s);
    }
    invh[k] = inv;
  }
  if (k < 64) {
    float inv = 0.0f;
    if (k < KL) {
      float a = wl[2 * k], b = wl[2 * k + 1];
      inv = 1.0f / sqrtf(a * a + b * b);
    }
    invl[k] = inv;
  }
}

// Pack embed_high (f32, row-major 1000x78), PRE-SCALED by 1/||w||, into f16
// WMMA B-fragments. Fragment f = t*3+kc covers N-tile t (16 clusters) and
// K-chunk kc. Per lane: col = lane&15, kb = (lane<16)?0:8;
// e<8 -> K = kc*32+kb+e ; e>=8 -> K = kc*32+16+kb+(e-8).
__global__ __launch_bounds__(256) void prep_bfrag(const float* __restrict__ wh,
                                                  const float* __restrict__ invh,
                                                  _Float16* __restrict__ bfrag) {
  int i = blockIdx.x * 256 + threadIdx.x;
  if (i >= BFRAG_HALVES) return;
  int e    = i & 15;
  int lane = (i >> 4) & 31;
  int f    = i >> 9;
  int t    = f / 3;
  int kc   = f - t * 3;
  int cluster = t * 16 + (lane & 15);
  int kb   = (lane < 16) ? 0 : 8;
  int K    = (e < 8) ? (kc * 32 + kb + e) : (kc * 32 + 16 + kb + (e - 8));
  float v  = (cluster < KH && K < DH) ? wh[cluster * DH + K] * invh[cluster] : 0.0f;
  bfrag[i] = (_Float16)v;
}

__device__ __forceinline__ void load_btile(const _Float16* __restrict__ bp,
                                           int t, v16h b[3]) {
#pragma unroll
  for (int kc = 0; kc < 3; ++kc)
    b[kc] = *(const v16h*)(bp + (size_t)(t * 3 + kc) * 512);
}

__device__ __forceinline__ void mm_argmax(const v16h a[3], const v16h b[3],
                                          int col, float bestv[8], int besti[8]) {
  v8f c = {};
#pragma unroll
  for (int kc = 0; kc < 3; ++kc)
    c = __builtin_amdgcn_wmma_f32_16x16x32_f16(
        false, a[kc], false, b[kc], (short)0, c, false, false);
#pragma unroll
  for (int j = 0; j < 8; ++j)
    if (c[j] > bestv[j]) { bestv[j] = c[j]; besti[j] = col; }
}

// Fused f16-WMMA GEMM + row-wise argmax over 1000 clusters.
// 512 threads = 16 waves; each wave owns 16 rows; block owns 256 rows.
// Packed B (193.5 KB) is DMA'd once per block into LDS via the Tensor Data
// Mover and shared by all 16 waves; B-fragment reads are double-buffered in
// VGPRs so ds_load latency overlaps the WMMA chain.
__global__ __launch_bounds__(512) void gemm_argmax_high(
    const float* __restrict__ x, const _Float16* __restrict__ bfrag,
    int* __restrict__ idxh, float* __restrict__ out) {
  extern __shared__ _Float16 sB[];        // BFRAG_HALVES halves (dynamic LDS)
  __shared__ _Float16 sA[256 * DHP];      // 49152 B (static LDS)

  const int tid = threadIdx.x;
  const int rowBase = blockIdx.x * 256;

  // --- TDM: one DMA of the whole packed B, global -> LDS (1D tile of
  //     24192 x 8-byte elements). Descriptor per ISA 8.3-8.5. ---
  if (tid == 0) {
    unsigned int lds = (unsigned int)(uintptr_t)(&sB[0]);
    unsigned long long ga = (unsigned long long)(uintptr_t)bfrag;
    // group0: count=1 | lds_addr | global_addr[56:0] | type=2 (bits 127:126)
    v4u g0 = { 1u, lds, (unsigned int)ga,
               ((unsigned int)((ga >> 32) & 0x01FFFFFFull)) | 0x80000000u };
    // group1: data_size=3 (8B) ; tensor_dim0=24192 ; tensor_dim1=1 ;
    //         tile_dim0=24192 ; tile_dim1=1 ; tensor_dim0_stride=24192
    v8i g1 = { (int)0x00030000u,      // [17:16] data_size=3
               (int)0x5E800000u,      // [63:48] tensor_dim0.lo16 = 24192
               (int)(1 << 16),        // [95:80] tensor_dim1.lo16 = 1
               (int)0x5E800000u,      // [127:112] tile_dim0 = 24192
               1,                     // [143:128] tile_dim1 = 1
               24192,                 // [191:160] tensor_dim0_stride = 24192
               0, 0 };
    v4i g2 = { 0, 0, 0, 0 };
    v4i g3 = { 0, 0, 0, 0 };
#if defined(__clang_major__) && (__clang_major__ >= 23)
    v8i gz = { 0, 0, 0, 0, 0, 0, 0, 0 };
    __builtin_amdgcn_tensor_load_to_lds(g0, g1, g2, g3, gz, 0);
#else
    __builtin_amdgcn_tensor_load_to_lds(g0, g1, g2, g3, 0);
#endif
  }

  // Stage + f16-convert the A tile (256 rows x 96 K, zero-padded) while the
  // TDM copy is in flight.
  for (int i = tid; i < 256 * DHP; i += 512) {
    int r = i / DHP, k = i - r * DHP;
    float v = (k < DH) ? x[(size_t)(rowBase + r) * D_ + k] : 0.0f;
    sA[i] = (_Float16)v;
  }
  __builtin_amdgcn_s_wait_tensorcnt(0);
  __syncthreads();

  const int wave = tid >> 5;     // 0..15
  const int lane = tid & 31;
  const int lh   = lane & 15;
  const int kb   = (lane < 16) ? 0 : 8;
  const int r    = wave * 16 + lh;

  // Assemble 3 A fragments (K=0..31, 32..63, 64..95) per ISA 16-bit A layout.
  v16h a[3];
#pragma unroll
  for (int kc = 0; kc < 3; ++kc) {
    v8h lo = *(const v8h*)&sA[r * DHP + kc * 32 + kb];
    v8h hi = *(const v8h*)&sA[r * DHP + kc * 32 + 16 + kb];
    a[kc] = __builtin_shufflevector(lo, hi, 0, 1, 2, 3, 4, 5, 6, 7,
                                    8, 9, 10, 11, 12, 13, 14, 15);
  }

  float bestv[8];
  int   besti[8];
#pragma unroll
  for (int j = 0; j < 8; ++j) { bestv[j] = -INFINITY; besti[j] = 0; }

  const _Float16* bp = sB + (size_t)lane * 16;

  // Software-pipelined hot loop over fully-valid tiles 0..61 (cols < 992):
  // loads of tile t+1 are in flight while tile t's WMMAs execute.
  v16h b0[3], b1[3];
  load_btile(bp, 0, b0);
  for (int t = 0; t < NT - 1; t += 2) {
    load_btile(bp, t + 1, b1);
    mm_argmax(a, b0, t * 16 + lh, bestv, besti);
    load_btile(bp, t + 2, b0);          // t+2 <= 62 (last prefetch = tile 62)
    mm_argmax(a, b1, (t + 1) * 16 + lh, bestv, besti);
  }
  // Peeled final tile (t = 62) already resident in b0: cols 992..1007,
  // only first 8 valid.
  {
    v8f c = {};
#pragma unroll
    for (int kc = 0; kc < 3; ++kc)
      c = __builtin_amdgcn_wmma_f32_16x16x32_f16(
          false, a[kc], false, b0[kc], (short)0, c, false, false);
    int col = (NT - 1) * 16 + lh;
    if (col < KH) {
#pragma unroll
      for (int j = 0; j < 8; ++j) {
        if (c[j] > bestv[j]) { bestv[j] = c[j]; besti[j] = col; }
      }
    }
  }

  // Cross-lane argmax reduction within each 16-lane half (rows j / j+8).
#pragma unroll
  for (int j = 0; j < 8; ++j) {
#pragma unroll
    for (int off = 8; off > 0; off >>= 1) {
      float ov = __shfl_xor(bestv[j], off, 32);
      int   oi = __shfl_xor(besti[j], off, 32);
      if (ov > bestv[j] || (ov == bestv[j] && oi < besti[j])) {
        bestv[j] = ov; besti[j] = oi;
      }
    }
    if (lh == 0) {
      int M = (lane < 16) ? j : (j + 8);
      int grow = rowBase + wave * 16 + M;
      idxh[grow] = besti[j];
      out[OUT_IDXH + grow] = (float)besti[j];
    }
  }
}

// One wave per (b,t) row: gather W[idx], f32 softmax over 78 dims, write anchor;
// lane 0 also does the tiny 50x2 low-frequency clustering + 2-dim softmax.
__global__ __launch_bounds__(256) void anchor_kernel(
    const float* __restrict__ x, const float* __restrict__ wh,
    const float* __restrict__ wl, const float* __restrict__ invl,
    const int* __restrict__ idxh, float* __restrict__ out) {
  const int wave = threadIdx.x >> 5;
  const int lane = threadIdx.x & 31;
  const int row  = blockIdx.x * 8 + wave;
  const float* xr = x + (size_t)row * D_;

  // ---- high part ----
  int ih = idxh[row];
  const float* wr = wh + (size_t)ih * DH;
  int d0 = lane, d1 = lane + 32, d2 = lane + 64;
  bool v2 = (d2 < DH);
  float w0 = wr[d0], w1 = wr[d1], w2 = v2 ? wr[d2] : 0.0f;
  float z0 = w0 * xr[d0];
  float z1 = w1 * xr[d1];
  float z2 = v2 ? (w2 * xr[d2]) : -INFINITY;

  float m = fmaxf(z0, fmaxf(z1, z2));
#pragma unroll
  for (int off = 16; off > 0; off >>= 1) m = fmaxf(m, __shfl_xor(m, off, 32));

  float e0 = __expf(z0 - m);
  float e1 = __expf(z1 - m);
  float e2 = v2 ? __expf(z2 - m) : 0.0f;
  float s = e0 + e1 + e2;
#pragma unroll
  for (int off = 16; off > 0; off >>= 1) s += __shfl_xor(s, off, 32);
  float rs = 1.0f / s;

  float* orow = out + (size_t)row * D_;
  orow[d0] = e0 * rs * w0;
  orow[d1] = e1 * rs * w1;
  if (v2) orow[d2] = e2 * rs * w2;

  // ---- low part (2 dims, 50 clusters) ----
  if (lane == 0) {
    float x0 = xr[DH], x1 = xr[DH + 1];
    float best = -INFINITY;
    int bi = 0;
    for (int k = 0; k < KL; ++k) {
      float sim = (x0 * wl[2 * k] + x1 * wl[2 * k + 1]) * invl[k];
      if (sim > best) { best = sim; bi = k; }
    }
    float a0 = wl[2 * bi], a1 = wl[2 * bi + 1];
    float t0 = a0 * x0, t1 = a1 * x1;
    float ml = fmaxf(t0, t1);
    float f0 = __expf(t0 - ml), f1 = __expf(t1 - ml);
    float rl = 1.0f / (f0 + f1);
    orow[DH]     = f0 * rl * a0;
    orow[DH + 1] = f1 * rl * a1;
    out[OUT_IDXL + row] = (float)bi;
  }
}

extern "C" void kernel_launch(void* const* d_in, const int* in_sizes, int n_in,
                              void* d_out, int out_size, void* d_ws, size_t ws_size,
                              hipStream_t stream) {
  (void)in_sizes; (void)n_in; (void)out_size; (void)ws_size;
  const float* x  = (const float*)d_in[0];   // (32,4096,80) f32
  const float* wh = (const float*)d_in[1];   // (1000,78) f32
  const float* wl = (const float*)d_in[2];   // (50,2) f32
  float* out = (float*)d_out;

  char* ws = (char*)d_ws;
  _Float16* bfrag = (_Float16*)(ws + WS_BFRAG);
  float*    invh  = (float*)(ws + WS_INVH);
  float*    invl  = (float*)(ws + WS_INVL);
  int*      idxh  = (int*)(ws + WS_IDXH);

  prep_norms<<<(KHP + 255) / 256, 256, 0, stream>>>(wh, wl, invh, invl);
  prep_bfrag<<<(BFRAG_HALVES + 255) / 256, 256, 0, stream>>>(wh, invh, bfrag);
  gemm_argmax_high<<<NROWS / 256, 512, BFRAG_BYTES, stream>>>(x, bfrag, idxh, out);
  anchor_kernel<<<NROWS / 8, 256, 0, stream>>>(x, wh, wl, invl, idxh, out);
}
